// MultiheadAttention_3161095930000
// MI455X (gfx1250) — matrix operand order repaired and checked by reading, not compiled, not run on hardware
//
#include <hip/hip_runtime.h>
#include <hip/hip_bf16.h>
#include <cstdint>

typedef __attribute__((ext_vector_type(16))) _Float16 v16h;
typedef __attribute__((ext_vector_type(8)))  _Float16 v8h;
typedef __attribute__((ext_vector_type(8)))  float    v8f;

#define BATCH 4
#define NCTX  2048
#define WIDTH 1024
#define HEADS 16
#define HDIM  64
#define MTOT  (BATCH * NCTX)            // 8192
#define QKSCALE 0.35355339059327373f    // 64^-0.25 applied to both q and k

__device__ __forceinline__ int lane_id() { return (int)(threadIdx.x & 31); }

__device__ __forceinline__ v16h make_frag(const _Float16* p0, const _Float16* p1) {
  v8h lo = *(const v8h*)p0;
  v8h hi = *(const v8h*)p1;
  v16h f;
#pragma unroll
  for (int i = 0; i < 8; ++i) { f[i] = lo[i]; f[8 + i] = hi[i]; }
  return f;
}

// A fragment (16x32 f16), row-major source, leading dim ld.
// Lane l: row = l&15, K-half = l>>4. Elements 0-7 <-> K=kh*8.., 8-15 <-> K=16+kh*8..
__device__ __forceinline__ v16h load_a_frag(const _Float16* base, int ld) {
  int l = lane_id();
  const _Float16* p = base + (l & 15) * ld + (l >> 4) * 8;
  return make_frag(p, p + 16);
}

// B fragment (32x16 f16) from K-contiguous (pre-transposed) source:
// column n is stored as row n of baseT; elements 0-7 <-> K=kh*8.., 8-15 <-> K=16+kh*8..
__device__ __forceinline__ v16h load_b_frag(const _Float16* baseT, int ld) {
  int l = lane_id();
  const _Float16* p = baseT + (l & 15) * ld + (l >> 4) * 8;
  return make_frag(p, p + 16);
}

__device__ __forceinline__ v8f wmma_f32(v16h a, v16h b, v8f c) {
  return __builtin_amdgcn_wmma_f32_16x16x32_f16(false, a, false, b, (short)0, c,
                                                false, false);
}

// gfx1250 async copy: 16B global -> LDS, tracked by ASYNCcnt.
__device__ __forceinline__ void async_copy_b128(const _Float16* g, _Float16* l) {
  unsigned lds = (unsigned)(uintptr_t)l;  // flat LDS pointer: addr[31:0] = LDS offset
  asm volatile("global_load_async_to_lds_b128 %0, %1, off"
               :: "v"(lds), "v"(g) : "memory");
}
__device__ __forceinline__ void wait_async0() {
  asm volatile("s_wait_asynccnt 0x0" ::: "memory");
}

// ---------------- prep kernels ----------------
__global__ void cvt_f32_f16(const float* __restrict__ in, _Float16* __restrict__ out,
                            int n) {
  int i = blockIdx.x * blockDim.x + threadIdx.x;
  if (i < n) out[i] = (_Float16)in[i];
}

// w [K, Nout] f32 row-major  ->  wT [Nout, K] f16 (K contiguous)
__global__ void transpose_f32_f16(const float* __restrict__ w, _Float16* __restrict__ wT,
                                  int K, int Nout) {
  int i = blockIdx.x * blockDim.x + threadIdx.x;
  if (i < K * Nout) {
    int o = i / K;
    int k = i - o * K;
    wT[i] = (_Float16)w[k * Nout + o];
  }
}

// ---------------- QKV GEMM: [8192,1024] x [1024,3072] ----------------
// Wave computes a 32x64 tile (2 row-tiles share the B fragments -> 8 WMMA per
// 12 b128 loads). Epilogue: +bias, fold qk scale, scatter to q/k [b,h,n,d] and
// transposed v [b,h,d,n] (f16).
__global__ __launch_bounds__(256) void qkv_gemm(
    const _Float16* __restrict__ xh, const _Float16* __restrict__ wT,
    const float* __restrict__ bias, _Float16* __restrict__ qh,
    _Float16* __restrict__ kh, _Float16* __restrict__ vT) {
  int wave = threadIdx.x >> 5;
  int l = lane_id();
  int half = l >> 4, n16 = l & 15;
  int m0 = blockIdx.y * 256 + wave * 32;
  int c0 = blockIdx.x * 64;

  v8f acc[2][4] = {};
  const _Float16* arow = xh + (size_t)m0 * WIDTH;
#pragma unroll 2
  for (int k0 = 0; k0 < WIDTH; k0 += 32) {
    __builtin_prefetch(arow + k0 + 256, 0, 1);
    v16h a0 = load_a_frag(arow + k0, WIDTH);
    v16h a1 = load_a_frag(arow + 16 * WIDTH + k0, WIDTH);
#pragma unroll
    for (int t = 0; t < 4; ++t) {
      v16h b = load_b_frag(wT + (size_t)(c0 + t * 16) * WIDTH + k0, WIDTH);
      acc[0][t] = wmma_f32(a0, b, acc[0][t]);
      acc[1][t] = wmma_f32(a1, b, acc[1][t]);
    }
  }

#pragma unroll
  for (int rt = 0; rt < 2; ++rt) {
#pragma unroll
    for (int t = 0; t < 4; ++t) {
      int j = c0 + t * 16 + n16;        // channel 0..3071
      int h = j / 192;
      int w = j - h * 192;
      int ty = w >> 6;                   // 0=q, 1=k, 2=v
      int d = w & 63;
      float bj = bias[j];
#pragma unroll
      for (int r = 0; r < 8; ++r) {
        int m = m0 + rt * 16 + r + 8 * half;
        int bidx = m >> 11;              // / NCTX
        int nidx = m & (NCTX - 1);
        int bh = bidx * HEADS + h;
        float v = acc[rt][t][r] + bj;
        if (ty == 0) {
          qh[((size_t)bh * NCTX + nidx) * HDIM + d] = (_Float16)(v * QKSCALE);
        } else if (ty == 1) {
          kh[((size_t)bh * NCTX + nidx) * HDIM + d] = (_Float16)(v * QKSCALE);
        } else {
          vT[((size_t)bh * HDIM + d) * NCTX + nidx] = (_Float16)v;
        }
      }
    }
  }
}

// ---------------- flash attention ----------------
// grid (B*H, NCTX/128); each wave owns 16 query rows. All 8 waves stream the
// same 64-key block, so K and V^T tiles are staged in LDS once per block via
// gfx1250 async copies (double-buffered under the WMMA work).
__global__ __launch_bounds__(256) void flash_attn(
    const _Float16* __restrict__ qh, const _Float16* __restrict__ kh,
    const _Float16* __restrict__ vT, _Float16* __restrict__ attn) {
  __shared__ _Float16 pbuf[8][16 * 72];   // per-wave P tile (C->A relayout)
  __shared__ _Float16 kbuf[2][64 * 72];   // keys x d, padded stride 72
  __shared__ _Float16 vbuf[2][64 * 72];   // d x keys, padded stride 72
  int tid = threadIdx.x;
  int wave = tid >> 5;
  int l = lane_id();
  int half = l >> 4, n16 = l & 15;
  int bh = blockIdx.x;
  int b = bh >> 4, h = bh & 15;
  int q0 = blockIdx.y * 128 + wave * 16;

  const _Float16* qb = qh + (size_t)bh * NCTX * HDIM;
  const _Float16* kb = kh + (size_t)bh * NCTX * HDIM;
  const _Float16* vb = vT + (size_t)bh * HDIM * NCTX;

  v16h qa[2];
#pragma unroll
  for (int kk = 0; kk < 2; ++kk)
    qa[kk] = load_a_frag(qb + (size_t)q0 * HDIM + kk * 32, HDIM);

  v8f o[4] = {};
  float mrow[8], lrow[8];
#pragma unroll
  for (int r = 0; r < 8; ++r) { mrow[r] = -1e30f; lrow[r] = 0.f; }

  _Float16* pw = pbuf[wave];

  // async stage of one 64-key block: 512 x 16B chunks, 2 per thread per tensor
  auto stage = [&](int kb0, int s) {
#pragma unroll
    for (int u = 0; u < 2; ++u) {
      int c = tid * 2 + u;
      int row = c >> 3;            // 0..63
      int col = (c & 7) * 8;       // f16 units, 16B aligned
      async_copy_b128(kb + (size_t)(kb0 + row) * HDIM + col,
                      &kbuf[s][row * 72 + col]);
      async_copy_b128(vb + (size_t)row * NCTX + kb0 + col,
                      &vbuf[s][row * 72 + col]);
    }
  };

  stage(0, 0);
  wait_async0();
  __syncthreads();

  for (int kb0 = 0; kb0 < NCTX; kb0 += 64) {
    int ibuf = (kb0 >> 6) & 1;
    if (kb0 + 64 < NCTX) stage(kb0 + 64, ibuf ^ 1);  // prefetch next block

    // S = Q K^T for 16 queries x 64 keys (B frags from LDS)
    v8f s[4] = {};
#pragma unroll
    for (int t = 0; t < 4; ++t) {
#pragma unroll
      for (int kk = 0; kk < 2; ++kk) {
        v16h bkf = load_b_frag(&kbuf[ibuf][(t * 16) * 72 + kk * 32], 72);
        s[t] = wmma_f32(qa[kk], bkf, s[t]);
      }
    }
    // online softmax; C-layout row = r + 8*half, replicated across 16-lane half
#pragma unroll
    for (int r = 0; r < 8; ++r) {
      float mx = fmaxf(fmaxf(s[0][r], s[1][r]), fmaxf(s[2][r], s[3][r]));
#pragma unroll
      for (int off = 8; off >= 1; off >>= 1)
        mx = fmaxf(mx, __shfl_xor(mx, off, 32));
      float newm = fmaxf(mrow[r], mx);
      float alpha = __expf(mrow[r] - newm);
      float rs = 0.f;
#pragma unroll
      for (int t = 0; t < 4; ++t) {
        float e = __expf(s[t][r] - newm);
        s[t][r] = e;
        rs += e;
      }
#pragma unroll
      for (int off = 8; off >= 1; off >>= 1)
        rs += __shfl_xor(rs, off, 32);
      lrow[r] = lrow[r] * alpha + rs;
      mrow[r] = newm;
#pragma unroll
      for (int t = 0; t < 4; ++t) o[t][r] *= alpha;
    }
    // stage P (C-layout) into per-wave LDS, reload as A fragments
#pragma unroll
    for (int t = 0; t < 4; ++t)
#pragma unroll
      for (int r = 0; r < 8; ++r)
        pw[(r + 8 * half) * 72 + t * 16 + n16] = (_Float16)s[t][r];
    __builtin_amdgcn_wave_barrier();
#pragma unroll
    for (int kk = 0; kk < 2; ++kk) {
      const _Float16* p = pw + (l & 15) * 72 + kk * 32 + (l >> 4) * 8;
      v16h pa = make_frag(p, p + 16);
#pragma unroll
      for (int t = 0; t < 4; ++t) {
        v16h bvf = load_b_frag(&vbuf[ibuf][(t * 16) * 72 + kk * 32], 72);
        o[t] = wmma_f32(pa, bvf, o[t]);
      }
    }
    __builtin_amdgcn_wave_barrier();

    wait_async0();     // next block's tiles landed while we computed
    __syncthreads();   // everyone done reading this block, next buffer ready
  }

  // normalize and write attn in [b, n, h*64+d] f16 layout for the proj GEMM
#pragma unroll
  for (int r = 0; r < 8; ++r) {
    int qrow = q0 + r + 8 * half;
    float inv = 1.0f / lrow[r];
#pragma unroll
    for (int t = 0; t < 4; ++t)
      attn[((size_t)b * NCTX + qrow) * WIDTH + h * HDIM + t * 16 + n16] =
          (_Float16)(o[t][r] * inv);
  }
}

// ---------------- projection GEMM: [8192,1024] x [1024,1024] + bias -> f32 ----
__global__ __launch_bounds__(256) void proj_gemm(
    const _Float16* __restrict__ ah, const _Float16* __restrict__ wT,
    const float* __restrict__ bias, float* __restrict__ out) {
  int wave = threadIdx.x >> 5;
  int l = lane_id();
  int half = l >> 4, n16 = l & 15;
  int m0 = blockIdx.y * 256 + wave * 32;
  int c0 = blockIdx.x * 64;

  v8f acc[2][4] = {};
  const _Float16* arow = ah + (size_t)m0 * WIDTH;
#pragma unroll 2
  for (int k0 = 0; k0 < WIDTH; k0 += 32) {
    __builtin_prefetch(arow + k0 + 256, 0, 1);
    v16h a0 = load_a_frag(arow + k0, WIDTH);
    v16h a1 = load_a_frag(arow + 16 * WIDTH + k0, WIDTH);
#pragma unroll
    for (int t = 0; t < 4; ++t) {
      v16h b = load_b_frag(wT + (size_t)(c0 + t * 16) * WIDTH + k0, WIDTH);
      acc[0][t] = wmma_f32(a0, b, acc[0][t]);
      acc[1][t] = wmma_f32(a1, b, acc[1][t]);
    }
  }
#pragma unroll
  for (int rt = 0; rt < 2; ++rt) {
#pragma unroll
    for (int t = 0; t < 4; ++t) {
      int j = c0 + t * 16 + n16;
      float bj = bias[j];
#pragma unroll
      for (int r = 0; r < 8; ++r) {
        int m = m0 + rt * 16 + r + 8 * half;
        out[(size_t)m * WIDTH + j] = acc[rt][t][r] + bj;
      }
    }
  }
}

extern "C" void kernel_launch(void* const* d_in, const int* in_sizes, int n_in,
                              void* d_out, int out_size, void* d_ws, size_t ws_size,
                              hipStream_t stream) {
  const float* x      = (const float*)d_in[0];
  const float* w_qkv  = (const float*)d_in[1];
  const float* b_qkv  = (const float*)d_in[2];
  const float* w_proj = (const float*)d_in[3];
  const float* b_proj = (const float*)d_in[4];
  float* out = (float*)d_out;

  char* ws = (char*)d_ws;
  size_t off = 0;
  auto carve = [&](size_t bytes) {
    void* p = ws + off;
    off += (bytes + 255) & ~(size_t)255;
    return p;
  };
  const size_t EL = (size_t)MTOT * WIDTH;  // 8Mi elements
  _Float16* xh     = (_Float16*)carve(EL * 2);
  _Float16* wqkvT  = (_Float16*)carve((size_t)3 * WIDTH * WIDTH * 2);
  _Float16* wprojT = (_Float16*)carve((size_t)WIDTH * WIDTH * 2);
  _Float16* qh     = (_Float16*)carve(EL * 2);
  _Float16* kh     = (_Float16*)carve(EL * 2);
  _Float16* vT     = (_Float16*)carve(EL * 2);
  _Float16* attn   = (_Float16*)carve(EL * 2);

  // prep: f32 -> f16 casts and weight transposes (K-contiguous B operands)
  {
    int n = (int)EL;
    cvt_f32_f16<<<(n + 255) / 256, 256, 0, stream>>>(x, xh, n);
    int nq = 3 * WIDTH * WIDTH;
    transpose_f32_f16<<<(nq + 255) / 256, 256, 0, stream>>>(w_qkv, wqkvT, WIDTH,
                                                            3 * WIDTH);
    int np = WIDTH * WIDTH;
    transpose_f32_f16<<<(np + 255) / 256, 256, 0, stream>>>(w_proj, wprojT, WIDTH,
                                                            WIDTH);
  }
  // qkv = x @ w_qkv + b_qkv, scatter into q/k (scaled) and transposed v
  qkv_gemm<<<dim3(3 * WIDTH / 64, MTOT / 256), 256, 0, stream>>>(xh, wqkvT, b_qkv, qh,
                                                                 kh, vT);
  // flash attention per (b, h)
  flash_attn<<<dim3(BATCH * HEADS, NCTX / 128), 256, 0, stream>>>(qh, kh, vT, attn);
  // out = attn @ w_proj + b_proj
  proj_gemm<<<dim3(WIDTH / 64, MTOT / 256), 256, 0, stream>>>(attn, wprojT, b_proj,
                                                              out);
}
